// Astormer_72576357368451
// MI455X (gfx1250) — matrix operand order repaired and checked by reading, not compiled, not run on hardware
//
#include <hip/hip_runtime.h>
#include <hip/hip_bf16.h>

// Model dims (fixed by the reference)
#define Hd   512
#define NHd  8
#define DHd  64
#define NLd  2
#define RRd  64
#define FFd  2048
#define Bd   4
#define Td   512
#define Sd   512
#define NEGV (-1e10f)
#define EPSV (1e-5f)

typedef __attribute__((ext_vector_type(16))) _Float16 v16h;
typedef __attribute__((ext_vector_type(8)))  float    v8f;
typedef __attribute__((ext_vector_type(4)))  unsigned int v4u;
typedef __attribute__((ext_vector_type(2)))  float    f2v;
typedef __attribute__((ext_vector_type(2)))  __fp16   fp16x2;

union FragH { v16h v; unsigned int u[8]; v4u q[2]; };

#define TILE_M 256
#define TILE_N 64
#define TILE_K 32

static __device__ inline unsigned int pack_h2(float a, float b) {
#if __has_builtin(__builtin_amdgcn_cvt_pkrtz)
    union { fp16x2 h; unsigned int u; } c;
    c.h = __builtin_amdgcn_cvt_pkrtz(a, b);
    return c.u;
#else
    union { _Float16 h[2]; unsigned int u; } c;
    c.h[0] = (_Float16)a; c.h[1] = (_Float16)b; return c.u;
#endif
}

// ---------------------------------------------------------------------------
// Batched strided GEMM: C = A * B (or A * B^T), f16 WMMA, f32 accumulate.
// LDS tiles are stored PRE-SWIZZLED in WMMA fragment layout so each lane
// fetches a fragment with two ds_load_b128.
//   A fragment layout (16x32 f16): k = (v>=4?16:0) + (lane>=16?8:0) + 2*(v&3)
//     -> k-pair p: lane = (r&15) + 16*((p>>2)&1), vgpr = ((p>>3)<<2)|(p&3)
//   B fragment layout (32x16 f16): k = (v>=4?16:0) + (lane>=16?8:0) + 2*(v&3)
//     -> k-pair p: lane = (n&15) + 16*((p>>2)&1), vgpr = ((p>>3)<<2)|(p&3)
// epi: 0=store  1=store+bias  2=store+bias+relu  3=accumulate (C += r)
// Requires M%256==0, N%64==0, K%32==0 (true for all shapes here).
// ---------------------------------------------------------------------------
__global__ __launch_bounds__(256)
void wmma_gemm_kernel(const float* __restrict__ A, const float* __restrict__ Bm,
                      const float* __restrict__ bias, float* __restrict__ C,
                      int M, int N, int K,
                      int lda, int ldb, int ldc,
                      long sAb, long sAh, long sBb, long sBh, long sCb, long sCh,
                      int nh, int transB, int epi)
{
    // 16 row-tiles * 32 lanes * 8 dwords = 4096 dwords (16KB)
    __shared__ __align__(16) unsigned int lA[16 * 32 * 8];
    // 4 col-tiles * 32 lanes * 8 dwords = 1024 dwords (4KB)
    __shared__ __align__(16) unsigned int lB[4 * 32 * 8];

    const int z  = blockIdx.z;
    const int bb = z / nh, hh = z % nh;
    A  += bb * sAb + hh * sAh;
    Bm += bb * sBb + hh * sBh;
    C  += bb * sCb + hh * sCh;

    const int tid  = threadIdx.x;
    const int lane = tid & 31;
    const int w    = tid >> 5;          // wave id 0..7, covers rows [32w, 32w+32)
    const int row0 = blockIdx.y * TILE_M;
    const int col0 = blockIdx.x * TILE_N;

    v8f zero = {0.f,0.f,0.f,0.f,0.f,0.f,0.f,0.f};
    v8f acc[2][4];
    #pragma unroll
    for (int s = 0; s < 2; ++s)
        #pragma unroll
        for (int j = 0; j < 4; ++j) acc[s][j] = zero;

    for (int k0 = 0; k0 < K; k0 += TILE_K) {
        // ---- stage A tile (256 rows x 32 k), fp32 -> packed f16, swizzled ----
        #pragma unroll
        for (int i = 0; i < 16; ++i) {
            int idx = tid + 256 * i;
            int p = idx & 15, r = idx >> 4;            // k-pair, tile row
            f2v f = *(const f2v*)(A + (long)(row0 + r) * lda + (k0 + 2 * p));
            int lane_d = (r & 15) + (((p >> 2) & 1) << 4);
            int vd     = ((p >> 3) << 2) | (p & 3);
            lA[(((r >> 4) * 32 + lane_d) << 3) | vd] = pack_h2(f.x, f.y);
        }
        // ---- stage B tile (64 cols x 32 k), swizzled ----
        #pragma unroll
        for (int i = 0; i < 4; ++i) {
            int idx = tid + 256 * i;
            int p = idx & 15, n = idx >> 4;            // k-pair, tile col
            float f0, f1;
            if (transB) {
                f2v f = *(const f2v*)(Bm + (long)(col0 + n) * ldb + (k0 + 2 * p));
                f0 = f.x; f1 = f.y;
            } else {
                f0 = Bm[(long)(k0 + 2 * p)     * ldb + (col0 + n)];
                f1 = Bm[(long)(k0 + 2 * p + 1) * ldb + (col0 + n)];
            }
            int lane_d = (n & 15) + (((p >> 2) & 1) << 4);
            int vd     = ((p >> 3) << 2) | (p & 3);
            lB[(((n >> 4) * 32 + lane_d) << 3) | vd] = pack_h2(f0, f1);
        }
        // ---- prefetch next k-tile (global_prefetch_b8) ----
        if (k0 + TILE_K < K) {
            __builtin_prefetch(A + (long)(row0 + tid) * lda + (k0 + TILE_K));
            if (tid < TILE_N) {
                const float* pb = transB
                    ? (Bm + (long)(col0 + tid) * ldb + (k0 + TILE_K))
                    : (Bm + (long)(k0 + TILE_K + (tid & 31)) * ldb + col0);
                __builtin_prefetch(pb);
            }
        }
        __syncthreads();

        // ---- fragments: 2x ds_load_b128 each ----
        const v4u* lA4 = (const v4u*)lA;
        const v4u* lB4 = (const v4u*)lB;
        FragH fa0, fa1;
        int a0 = ((2 * w + 0) * 32 + lane) << 1;
        int a1 = ((2 * w + 1) * 32 + lane) << 1;
        fa0.q[0] = lA4[a0]; fa0.q[1] = lA4[a0 + 1];
        fa1.q[0] = lA4[a1]; fa1.q[1] = lA4[a1 + 1];
        #pragma unroll
        for (int j = 0; j < 4; ++j) {
            FragH fb;
            int b0 = (j * 32 + lane) << 1;
            fb.q[0] = lB4[b0]; fb.q[1] = lB4[b0 + 1];
            acc[0][j] = __builtin_amdgcn_wmma_f32_16x16x32_f16(
                false, fa0.v, false, fb.v, (short)0, acc[0][j], false, false);
            acc[1][j] = __builtin_amdgcn_wmma_f32_16x16x32_f16(
                false, fa1.v, false, fb.v, (short)0, acc[1][j], false, false);
        }
        __syncthreads();
    }

    // ---- epilogue: row = 32w + 16s + (lane>=16?8:0)+v, col = 16j + (lane&15) ----
    const int cbase  = col0 + (lane & 15);
    const int laneHi = lane >> 4;
    #pragma unroll
    for (int s = 0; s < 2; ++s) {
        int rb = row0 + (w << 5) + (s << 4) + (laneHi << 3);
        #pragma unroll
        for (int j = 0; j < 4; ++j) {
            int gc = cbase + (j << 4);
            float bv = (epi == 1 || epi == 2) ? bias[gc] : 0.f;
            #pragma unroll
            for (int v = 0; v < 8; ++v) {
                float x = acc[s][j][v] + bv;
                if (epi == 2) x = fmaxf(x, 0.f);
                long off = (long)(rb + v) * ldc + gc;
                if (epi == 3) C[off] += x; else C[off] = x;
            }
        }
    }
}

// ---------------------------------------------------------------------------
// Self-attn softmax: e[b,h,l,:] = softmax(((QK) + P[b,l,h,rid]) / scale, mask)
// mask = (rid != PAD) && (m <= l).  In-place on e.  Block per (b,h,l).
// ---------------------------------------------------------------------------
__global__ __launch_bounds__(256)
void softmax_self_kernel(float* __restrict__ e, const float* __restrict__ P,
                         const int* __restrict__ rel_ids, float inv_scale)
{
    int idx = blockIdx.x;
    int l = idx % Td; int h = (idx / Td) % NHd; int b = idx / (Td * NHd);
    float* row = e + (((long)(b * NHd + h) * Td + l) * Td);
    const float* Prow = P + (((long)(b * Td + l) * NHd + h) * RRd);
    const int* rrow = rel_ids + ((long)b * Td + l) * Td;

    __shared__ float red[256];
    int tid = threadIdx.x;
    float vals[2];
    float mx = -3.4e38f;
    #pragma unroll
    for (int i = 0; i < 2; ++i) {
        int m = tid + 256 * i;
        int rid = rrow[m];
        float v = (row[m] + Prow[rid]) * inv_scale;
        bool valid = (rid != 0) && (m <= l);
        v = valid ? v : NEGV;
        vals[i] = v;
        mx = fmaxf(mx, v);
    }
    red[tid] = mx; __syncthreads();
    for (int s = 128; s > 0; s >>= 1) { if (tid < s) red[tid] = fmaxf(red[tid], red[tid + s]); __syncthreads(); }
    mx = red[0]; __syncthreads();
    float sum = 0.f;
    #pragma unroll
    for (int i = 0; i < 2; ++i) { vals[i] = __expf(vals[i] - mx); sum += vals[i]; }
    red[tid] = sum; __syncthreads();
    for (int s = 128; s > 0; s >>= 1) { if (tid < s) red[tid] += red[tid + s]; __syncthreads(); }
    float inv = 1.0f / red[0];
    #pragma unroll
    for (int i = 0; i < 2; ++i) row[tid + 256 * i] = vals[i] * inv;
}

// Cross-attn softmax with encoder mask (bool, 1 byte).
__global__ __launch_bounds__(256)
void softmax_cross_kernel(float* __restrict__ e, const unsigned char* __restrict__ mask,
                          float inv_scale)
{
    int idx = blockIdx.x;
    int l = idx % Td; int h = (idx / Td) % NHd; int b = idx / (Td * NHd);
    float* row = e + (((long)(b * NHd + h) * Td + l) * Sd);
    const unsigned char* mrow = mask + (long)b * Sd;

    __shared__ float red[256];
    int tid = threadIdx.x;
    float vals[2];
    float mx = -3.4e38f;
    #pragma unroll
    for (int i = 0; i < 2; ++i) {
        int m = tid + 256 * i;
        float v = row[m] * inv_scale;
        v = mrow[m] ? v : NEGV;
        vals[i] = v;
        mx = fmaxf(mx, v);
    }
    red[tid] = mx; __syncthreads();
    for (int s = 128; s > 0; s >>= 1) { if (tid < s) red[tid] = fmaxf(red[tid], red[tid + s]); __syncthreads(); }
    mx = red[0]; __syncthreads();
    float sum = 0.f;
    #pragma unroll
    for (int i = 0; i < 2; ++i) { vals[i] = __expf(vals[i] - mx); sum += vals[i]; }
    red[tid] = sum; __syncthreads();
    for (int s = 128; s > 0; s >>= 1) { if (tid < s) red[tid] += red[tid + s]; __syncthreads(); }
    float inv = 1.0f / red[0];
    #pragma unroll
    for (int i = 0; i < 2; ++i) row[tid + 256 * i] = vals[i] * inv;
}

// ---------------------------------------------------------------------------
// Relation binning: S[b,l,h,r] = sum_m a[b,h,l,m] * (rel_ids[b,l,m]==r)
// (rel_emb_v[PAD]==0, so binning all m matches the reference einsum exactly.)
// ---------------------------------------------------------------------------
__global__ __launch_bounds__(256)
void relbin_kernel(const float* __restrict__ a, const int* __restrict__ rel_ids,
                   float* __restrict__ Srel)
{
    int idx = blockIdx.x;                 // b*T + l
    int l = idx % Td, b = idx / Td;
    __shared__ float bins[NHd * RRd];
    int tid = threadIdx.x;
    for (int i = tid; i < NHd * RRd; i += 256) bins[i] = 0.f;
    __syncthreads();
    const int* rrow = rel_ids + ((long)b * Td + l) * Td;
    for (int m = tid; m < Td; m += 256) {
        int rid = rrow[m];
        #pragma unroll
        for (int h = 0; h < NHd; ++h) {
            float av = a[(((long)(b * NHd + h) * Td) + l) * Td + m];
            atomicAdd(&bins[h * RRd + rid], av);
        }
    }
    __syncthreads();
    float* out = Srel + ((long)(b * Td + l)) * (NHd * RRd);
    for (int i = tid; i < NHd * RRd; i += 256) out[i] = bins[i];
}

// ---------------------------------------------------------------------------
// Fused residual + LayerNorm, in-place on `res`: res = LN(res + x) * g + b
// ---------------------------------------------------------------------------
__global__ __launch_bounds__(256)
void ln_residual_kernel(float* __restrict__ res, const float* __restrict__ x,
                        const float* __restrict__ g, const float* __restrict__ bta)
{
    long row = blockIdx.x;
    float* r = res + row * Hd;
    const float* xx = x + row * Hd;
    __shared__ float red[256];
    int tid = threadIdx.x;
    float t0 = r[tid] + xx[tid];
    float t1 = r[tid + 256] + xx[tid + 256];
    red[tid] = t0 + t1; __syncthreads();
    for (int s = 128; s > 0; s >>= 1) { if (tid < s) red[tid] += red[tid + s]; __syncthreads(); }
    float mean = red[0] * (1.0f / Hd); __syncthreads();
    float d0 = t0 - mean, d1 = t1 - mean;
    red[tid] = d0 * d0 + d1 * d1; __syncthreads();
    for (int s = 128; s > 0; s >>= 1) { if (tid < s) red[tid] += red[tid + s]; __syncthreads(); }
    float rstd = rsqrtf(red[0] * (1.0f / Hd) + EPSV);
    r[tid]       = d0 * rstd * g[tid]       + bta[tid];
    r[tid + 256] = d1 * rstd * g[tid + 256] + bta[tid + 256];
}

// ---------------------------------------------------------------------------
extern "C" void kernel_launch(void* const* d_in, const int* in_sizes, int n_in,
                              void* d_out, int out_size, void* d_ws, size_t ws_size,
                              hipStream_t stream)
{
    (void)in_sizes; (void)n_in; (void)out_size; (void)ws_size;
    const float* q         = (const float*)d_in[0];
    const float* kv        = (const float*)d_in[1];
    const float* rel_emb_k = (const float*)d_in[2];
    const float* rel_emb_v = (const float*)d_in[3];
    const float* Wq_s = (const float*)d_in[4];  const float* bq_s = (const float*)d_in[5];
    const float* Wk_s = (const float*)d_in[6];  const float* Wv_s = (const float*)d_in[7];
    const float* Wo_s = (const float*)d_in[8];  const float* bo_s = (const float*)d_in[9];
    const float* ln1g = (const float*)d_in[10]; const float* ln1b = (const float*)d_in[11];
    const float* Wq_c = (const float*)d_in[12]; const float* bq_c = (const float*)d_in[13];
    const float* Wk_c = (const float*)d_in[14]; const float* Wv_c = (const float*)d_in[15];
    const float* Wo_c = (const float*)d_in[16]; const float* bo_c = (const float*)d_in[17];
    const float* ln2g = (const float*)d_in[18]; const float* ln2b = (const float*)d_in[19];
    const float* W1   = (const float*)d_in[20]; const float* b1   = (const float*)d_in[21];
    const float* W2   = (const float*)d_in[22]; const float* b2   = (const float*)d_in[23];
    const float* ln3g = (const float*)d_in[24]; const float* ln3b = (const float*)d_in[25];
    const int* rel_ids = (const int*)d_in[26];
    const unsigned char* enc_mask = (const unsigned char*)d_in[27];

    // Workspace layout (floats). M1 = B*T*H = 1M floats.
    const size_t M1 = (size_t)Bd * Td * Hd;
    float* wsf = (float*)d_ws;
    float* o    = wsf;            // activations (in-place LN target)
    float* Qb   = wsf + 1 * M1;
    float* Kb   = wsf + 2 * M1;
    float* Vb   = wsf + 3 * M1;
    float* Pb   = wsf + 4 * M1;   // [B,T,NH,R]
    float* Sb   = wsf + 5 * M1;   // [B,T,NH,R]
    float* att  = wsf + 6 * M1;
    float* tmp  = wsf + 7 * M1;
    float* e    = wsf + 8 * M1;   // [B,NH,T,S] = 8M floats; FFN hidden reuses it
    float* hbuf = e;

    (void)hipMemcpyAsync(o, q, M1 * sizeof(float), hipMemcpyDeviceToDevice, stream);

    auto gemm = [&](const float* A, const float* Bm, const float* bias, float* C,
                    int M, int N, int K, int lda, int ldb, int ldc,
                    long sAb, long sAh, long sBb, long sBh, long sCb, long sCh,
                    int nb, int nh, int transB, int epi) {
        dim3 grid(N / TILE_N, M / TILE_M, nb * nh);
        wmma_gemm_kernel<<<grid, dim3(256), 0, stream>>>(
            A, Bm, bias, C, M, N, K, lda, ldb, ldc,
            sAb, sAh, sBb, sBh, sCb, sCh, nh, transB, epi);
    };

    const float inv_scale = 0.125f;  // 1/sqrt(DH)
    const long sTT = (long)Td * Sd;

    for (int i = 0; i < NLd; ++i) {
        const size_t oHH = (size_t)i * Hd * Hd, oH = (size_t)i * Hd;
        // ---- relation-aware masked self-attention ----
        gemm(o, Wq_s + oHH, bq_s + oH, Qb, Bd*Td, Hd, Hd, Hd, Hd, Hd, 0,0,0,0,0,0, 1,1, 0, 1);
        gemm(o, Wk_s + oHH, nullptr,   Kb, Bd*Td, Hd, Hd, Hd, Hd, Hd, 0,0,0,0,0,0, 1,1, 0, 0);
        gemm(o, Wv_s + oHH, nullptr,   Vb, Bd*Td, Hd, Hd, Hd, Hd, Hd, 0,0,0,0,0,0, 1,1, 0, 0);
        // P[b,l,h,r] = Q . rel_emb_k^T   ([16384,64] x [64,64]^T)
        gemm(Qb, rel_emb_k, nullptr, Pb, Bd*Td*NHd, RRd, DHd, DHd, DHd, RRd, 0,0,0,0,0,0, 1,1, 1, 0);
        // e = Q K^T (batched per b,h)
        gemm(Qb, Kb, nullptr, e, Td, Sd, DHd, Hd, Hd, Sd,
             (long)Td*Hd, DHd, (long)Sd*Hd, DHd, (long)NHd*sTT, sTT, Bd, NHd, 1, 0);
        softmax_self_kernel<<<Bd*NHd*Td, 256, 0, stream>>>(e, Pb, rel_ids, inv_scale);
        // att = a V (batched)
        gemm(e, Vb, nullptr, att, Td, DHd, Sd, Sd, Hd, Hd,
             (long)NHd*sTT, sTT, (long)Sd*Hd, DHd, (long)Td*Hd, DHd, Bd, NHd, 0, 0);
        // att += (bin(a, rel_ids)) @ rel_emb_v
        relbin_kernel<<<Bd*Td, 256, 0, stream>>>(e, rel_ids, Sb);
        gemm(Sb, rel_emb_v, nullptr, att, Bd*Td*NHd, DHd, RRd, RRd, DHd, DHd, 0,0,0,0,0,0, 1,1, 0, 3);
        // output projection + residual LN
        gemm(att, Wo_s + oHH, bo_s + oH, tmp, Bd*Td, Hd, Hd, Hd, Hd, Hd, 0,0,0,0,0,0, 1,1, 0, 1);
        ln_residual_kernel<<<Bd*Td, 256, 0, stream>>>(o, tmp, ln1g + oH, ln1b + oH);

        // ---- cross-attention ----
        gemm(o,  Wq_c + oHH, bq_c + oH, Qb, Bd*Td, Hd, Hd, Hd, Hd, Hd, 0,0,0,0,0,0, 1,1, 0, 1);
        gemm(kv, Wk_c + oHH, nullptr,   Kb, Bd*Sd, Hd, Hd, Hd, Hd, Hd, 0,0,0,0,0,0, 1,1, 0, 0);
        gemm(kv, Wv_c + oHH, nullptr,   Vb, Bd*Sd, Hd, Hd, Hd, Hd, Hd, 0,0,0,0,0,0, 1,1, 0, 0);
        gemm(Qb, Kb, nullptr, e, Td, Sd, DHd, Hd, Hd, Sd,
             (long)Td*Hd, DHd, (long)Sd*Hd, DHd, (long)NHd*sTT, sTT, Bd, NHd, 1, 0);
        softmax_cross_kernel<<<Bd*NHd*Td, 256, 0, stream>>>(e, enc_mask, inv_scale);
        gemm(e, Vb, nullptr, att, Td, DHd, Sd, Sd, Hd, Hd,
             (long)NHd*sTT, sTT, (long)Sd*Hd, DHd, (long)Td*Hd, DHd, Bd, NHd, 0, 0);
        gemm(att, Wo_c + oHH, bo_c + oH, tmp, Bd*Td, Hd, Hd, Hd, Hd, Hd, 0,0,0,0,0,0, 1,1, 0, 1);
        ln_residual_kernel<<<Bd*Td, 256, 0, stream>>>(o, tmp, ln2g + oH, ln2b + oH);

        // ---- feed-forward ----
        gemm(o, W1 + (size_t)i*Hd*FFd, b1 + (size_t)i*FFd, hbuf,
             Bd*Td, FFd, Hd, Hd, FFd, FFd, 0,0,0,0,0,0, 1,1, 0, 2);
        gemm(hbuf, W2 + (size_t)i*FFd*Hd, b2 + oH, tmp,
             Bd*Td, Hd, FFd, FFd, Hd, Hd, 0,0,0,0,0,0, 1,1, 0, 1);
        ln_residual_kernel<<<Bd*Td, 256, 0, stream>>>(o, tmp, ln3g + oH, ln3b + oH);
    }

    (void)hipMemcpyAsync(d_out, o, M1 * sizeof(float), hipMemcpyDeviceToDevice, stream);
}
